// Block_82076825027291
// MI455X (gfx1250) — matrix operand order repaired and checked by reading, not compiled, not run on hardware
//
#include <hip/hip_runtime.h>
#include <hip/hip_bf16.h>
#include <math.h>

// ---------------------------------------------------------------------------
// Types for CDNA5 WMMA (wave32): v_wmma_f32_16x16x32_bf16
// ---------------------------------------------------------------------------
typedef __attribute__((ext_vector_type(16))) __bf16 bf16x16;
typedef __attribute__((ext_vector_type(8)))  __bf16 bf16x8;
typedef __attribute__((ext_vector_type(8)))  float  f32x8;
typedef __attribute__((ext_vector_type(4)))  int    i32x4;

__device__ inline bf16x8 lds_ld8(const __bf16* p) {
  return *reinterpret_cast<const bf16x8*>(p);
}
__device__ inline bf16x16 cat8(bf16x8 a, bf16x8 b) {
  return __builtin_shufflevector(a, b, 0,1,2,3,4,5,6,7,8,9,10,11,12,13,14,15);
}
__device__ inline f32x8 wmma_bf16(bf16x16 a, bf16x16 b, f32x8 c) {
  // D = A(16x32 bf16) * B(32x16 bf16) + C(16x16 f32)
  return __builtin_amdgcn_wmma_f32_16x16x32_bf16(false, a, false, b, (short)0, c,
                                                 false, false);
}

// ---------------------------------------------------------------------------
// Async global->LDS staging (gfx1250 GLOBAL_LOAD_ASYNC_TO_LDS_B128, ASYNCcnt)
// with a synchronous vector-copy fallback if the builtin is unavailable.
// Probe result: builtin exists; param 0 is AS1 (global) int4*, so the order is
// (global src, lds dst, imm offset, imm cpol).
// ---------------------------------------------------------------------------
#if defined(__has_builtin)
# if __has_builtin(__builtin_amdgcn_global_load_async_to_lds_b128) && \
     __has_builtin(__builtin_amdgcn_s_wait_asynccnt)
#  define USE_ASYNC_LDS 1
# endif
#endif
#ifndef USE_ASYNC_LDS
# define USE_ASYNC_LDS 0
#endif

#if USE_ASYNC_LDS
typedef __attribute__((address_space(1))) i32x4* gbl_i32x4_p;
typedef __attribute__((address_space(3))) i32x4* lds_i32x4_p;
#endif

__device__ inline void cp_b128(void* lds, const void* g) {
#if USE_ASYNC_LDS
  __builtin_amdgcn_global_load_async_to_lds_b128(
      (gbl_i32x4_p)(g), (lds_i32x4_p)(lds), 0, 0);
#else
  *reinterpret_cast<bf16x8*>(lds) = *reinterpret_cast<const bf16x8*>(g);
#endif
}
__device__ inline void cp_wait() {
#if USE_ASYNC_LDS
  __builtin_amdgcn_s_wait_asynccnt(0);
#endif
}

// ---------------------------------------------------------------------------
// f32 -> bf16 conversion (weights)
// ---------------------------------------------------------------------------
__global__ void f32_to_bf16_kernel(const float* __restrict__ in,
                                   __bf16* __restrict__ out, int n) {
  int i = blockIdx.x * blockDim.x + threadIdx.x;
  if (i < n) out[i] = (__bf16)in[i];
}

// ---------------------------------------------------------------------------
// LayerNorm (f32 in) -> bf16 out.  One block per row, C == 1024.
// ---------------------------------------------------------------------------
__global__ __launch_bounds__(256) void layernorm_to_bf16(
    const float* __restrict__ X, const float* __restrict__ W,
    __bf16* __restrict__ Hout, int C) {
  const int row = blockIdx.x;
  const int tid = threadIdx.x;
  const float* x = X + (size_t)row * C;
  float v[4];
  float s = 0.f;
#pragma unroll
  for (int i = 0; i < 4; ++i) { v[i] = x[tid + i * 256]; s += v[i]; }
  __shared__ float red[8];
#pragma unroll
  for (int off = 16; off; off >>= 1) s += __shfl_xor(s, off, 32);
  if ((tid & 31) == 0) red[tid >> 5] = s;
  __syncthreads();
  float mu = 0.f;
#pragma unroll
  for (int i = 0; i < 8; ++i) mu += red[i];
  mu /= (float)C;
  __syncthreads();
  float vs = 0.f;
#pragma unroll
  for (int i = 0; i < 4; ++i) { float d = v[i] - mu; vs += d * d; }
#pragma unroll
  for (int off = 16; off; off >>= 1) vs += __shfl_xor(vs, off, 32);
  if ((tid & 31) == 0) red[tid >> 5] = vs;
  __syncthreads();
  float var = 0.f;
#pragma unroll
  for (int i = 0; i < 8; ++i) var += red[i];
  var /= (float)C;
  const float rs = rsqrtf(var + 1e-5f);
#pragma unroll
  for (int i = 0; i < 4; ++i) {
    int c = tid + i * 256;
    Hout[(size_t)row * C + c] = (__bf16)((v[i] - mu) * rs * W[c]);
  }
}

// ---------------------------------------------------------------------------
// Tiled bf16 GEMM with WMMA.  C[M,N] = A[M,K] * B[K,N].
// Block tile 128x128, K step 32, 256 threads (8 waves, wave32).
// Wave w: rows (w&3)*32 .. +31 (2 row tiles), cols (w>>2)*64 .. +63 (4 col
// tiles) -> 8 WMMAs per staging round, acc[2][4] f32x8.
// EPI: 0 = store f32, 1 = +residual f32, 2 = ReLU -> bf16,
//      3 = split QKV into per-head [B,H,T,64] bf16 buffers.
// ---------------------------------------------------------------------------
template <int EPI>
__global__ __launch_bounds__(256) void gemm_bf16_wmma(
    const __bf16* __restrict__ A, const __bf16* __restrict__ Bw,
    int M, int N, int K,
    float* __restrict__ outF, const float* __restrict__ resid,
    __bf16* __restrict__ outB,
    __bf16* __restrict__ qb, __bf16* __restrict__ kb, __bf16* __restrict__ vb,
    int T, int C) {
  __shared__ __bf16 As[128][32];  // [m][k]
  __shared__ __bf16 Bs[128][32];  // [n][k]  (B tile transposed)
  const int tid = threadIdx.x;
  const int lane = tid & 31;
  const int wave = tid >> 5;
  const int laneM = lane & 15;
  const int half = lane >> 4;
  const int m0 = blockIdx.y * 128;
  const int n0 = blockIdx.x * 128;
  const int wr = wave & 3;   // row pair: rows wr*32 .. +31
  const int wc = wave >> 2;  // col quad: cols wc*64 .. +63

  f32x8 acc[2][4] = {};

  // A staging: 128 rows x 32 k = 4096 elems; 16 per thread (2 x b128, async)
  const int arow = tid >> 1, acol = (tid & 1) * 16;
  // B staging: 32 k-rows x 128 n; 16 per thread, transposed scatter
  const int bk = tid >> 3, bn = (tid & 7) * 16;

  for (int k0 = 0; k0 < K; k0 += 32) {
    __syncthreads();
    // Async-stage A tile (row-major, no transpose needed)
    cp_b128(&As[arow][acol], &A[(size_t)(m0 + arow) * K + k0 + acol]);
    cp_b128(&As[arow][acol + 8], &A[(size_t)(m0 + arow) * K + k0 + acol + 8]);
    // Stage B tile transposed: global row k, cols n -> Bs[n][k]
    bf16x8 br0 =
        *reinterpret_cast<const bf16x8*>(&Bw[(size_t)(k0 + bk) * N + n0 + bn]);
    bf16x8 br1 = *reinterpret_cast<const bf16x8*>(
        &Bw[(size_t)(k0 + bk) * N + n0 + bn + 8]);
#pragma unroll
    for (int i = 0; i < 8; ++i) {
      Bs[bn + i][bk] = br0[i];
      Bs[bn + 8 + i][bk] = br1[i];
    }
    cp_wait();
    __syncthreads();

    // Fragments: A (M = wr*32 + i*16 + laneM, k run 16*half .. +15),
    //            B (N = wc*64 + j*16 + laneM, k run 16*half .. +15)
    bf16x16 af[2], bfr[4];
#pragma unroll
    for (int i = 0; i < 2; ++i)
      af[i] = cat8(lds_ld8(&As[wr * 32 + i * 16 + laneM][16 * half]),
                   lds_ld8(&As[wr * 32 + i * 16 + laneM][16 * half + 8]));
#pragma unroll
    for (int j = 0; j < 4; ++j)
      bfr[j] = cat8(lds_ld8(&Bs[wc * 64 + j * 16 + laneM][16 * half]),
                    lds_ld8(&Bs[wc * 64 + j * 16 + laneM][16 * half + 8]));
#pragma unroll
    for (int i = 0; i < 2; ++i)
#pragma unroll
      for (int j = 0; j < 4; ++j) acc[i][j] = wmma_bf16(af[i], bfr[j], acc[i][j]);
  }

#pragma unroll
  for (int i = 0; i < 2; ++i) {
#pragma unroll
    for (int j = 0; j < 4; ++j) {
      const int ncol = n0 + wc * 64 + j * 16 + laneM;
#pragma unroll
      for (int r = 0; r < 8; ++r) {
        const int mrow = m0 + wr * 32 + i * 16 + r + 8 * half;  // M = r+8*half
        float val = acc[i][j][r];
        if constexpr (EPI == 0) {
          outF[(size_t)mrow * N + ncol] = val;
        } else if constexpr (EPI == 1) {
          outF[(size_t)mrow * N + ncol] = val + resid[(size_t)mrow * N + ncol];
        } else if constexpr (EPI == 2) {
          outB[(size_t)mrow * N + ncol] = (__bf16)(val > 0.f ? val : 0.f);
        } else {  // EPI == 3: scatter QKV to per-head layout
          const int which = ncol / C;
          const int c = ncol - which * C;
          const int h = c >> 6, d = c & 63;
          const int b = mrow / T, t = mrow - b * T;
          __bf16* dst = which == 0 ? qb : (which == 1 ? kb : vb);
          dst[((size_t)(b * (C >> 6) + h) * T + t) * 64 + d] = (__bf16)val;
        }
      }
    }
  }
}

// ---------------------------------------------------------------------------
// Causal flash attention, Dh = 64, bf16 WMMA.
// Grid: (B*H, T/64); block: 128 (4 waves). Each wave owns a 16-row query tile.
// K chunk (32 keys, async-staged) and V^T staged in LDS, shared by all waves.
// ---------------------------------------------------------------------------
__global__ __launch_bounds__(128) void attn_wmma(
    const __bf16* __restrict__ Q, const __bf16* __restrict__ Kb,
    const __bf16* __restrict__ Vb, __bf16* __restrict__ Y, int T, int H) {
  __shared__ __bf16 Ks[32][64];     // [key][dh]  == B-layout [n][k] for scores
  __shared__ __bf16 Vt[64][32];     // [dh][key]  == B-layout [n][k] for P*V
  __shared__ __bf16 Ps[4][16][32];  // per-wave P tile (C-layout -> A-layout)
  const int tid = threadIdx.x;
  const int lane = tid & 31;
  const int wave = tid >> 5;
  const int laneM = lane & 15;
  const int half = lane >> 4;
  const int bh = blockIdx.x;
  const int b = bh / H, h = bh - b * H;
  const int qt = blockIdx.y * 64 + wave * 16;
  const size_t base = (size_t)bh * T * 64;

  // Q fragments (A-layout), Dh split into K=32 chunks {0..31, 32..63}
  bf16x16 aq0, aq1;
  {
    const __bf16* qp = Q + base + (size_t)(qt + laneM) * 64;
    aq0 = cat8(*(const bf16x8*)(qp + 16 * half),
               *(const bf16x8*)(qp + 16 * half + 8));
    aq1 = cat8(*(const bf16x8*)(qp + 32 + 16 * half),
               *(const bf16x8*)(qp + 32 + 16 * half + 8));
  }

  f32x8 o[4] = {};
  float mrow[8], lrow[8];
#pragma unroll
  for (int r = 0; r < 8; ++r) { mrow[r] = -1e30f; lrow[r] = 0.f; }
  const float scale = 0.03125f;  // C**-0.5 = 1/32 (reference scales by full C)
  const int kEnd = (blockIdx.y + 1) * 64;

  for (int k0 = 0; k0 < kEnd; k0 += 32) {
    __syncthreads();
    {  // stage 32x64 K chunk (async) and transposed V chunk (128 threads)
      const int row = tid >> 2, cc = (tid & 3) * 16;
      const __bf16* kp = Kb + base + (size_t)(k0 + row) * 64 + cc;
      const __bf16* vp = Vb + base + (size_t)(k0 + row) * 64 + cc;
      cp_b128(&Ks[row][cc], kp);
      cp_b128(&Ks[row][cc + 8], kp + 8);
      bf16x8 v0 = *(const bf16x8*)(vp);
      bf16x8 v1 = *(const bf16x8*)(vp + 8);
#pragma unroll
      for (int i = 0; i < 8; ++i) {
        Vt[cc + i][row] = v0[i];
        Vt[cc + 8 + i][row] = v1[i];
      }
    }
    cp_wait();
    __syncthreads();

    // Scores for two 16-key subtiles: S = Q K^T (K-dim = Dh = 64 -> 2 WMMAs)
    f32x8 s[2];
#pragma unroll
    for (int sub = 0; sub < 2; ++sub) {
      f32x8 sa = {};
      bf16x16 b0 = cat8(lds_ld8(&Ks[sub * 16 + laneM][16 * half]),
                        lds_ld8(&Ks[sub * 16 + laneM][16 * half + 8]));
      bf16x16 b1 = cat8(lds_ld8(&Ks[sub * 16 + laneM][32 + 16 * half]),
                        lds_ld8(&Ks[sub * 16 + laneM][32 + 16 * half + 8]));
      sa = wmma_bf16(aq0, b0, sa);
      sa = wmma_bf16(aq1, b1, sa);
      s[sub] = sa;
    }

    // Online softmax across the 32 new keys (rows live in half-waves)
    float p[2][8], sf[8];
#pragma unroll
    for (int r = 0; r < 8; ++r) {
      const int qg = qt + r + 8 * half;
#pragma unroll
      for (int sub = 0; sub < 2; ++sub) {
        const int kg = k0 + sub * 16 + laneM;
        float sv = s[sub][r] * scale;
        if (kg > qg) sv = -1e30f;  // causal mask
        p[sub][r] = sv;
      }
      float lm = fmaxf(p[0][r], p[1][r]);
#pragma unroll
      for (int off = 1; off < 16; off <<= 1)
        lm = fmaxf(lm, __shfl_xor(lm, off, 16));
      const float mn = fmaxf(mrow[r], lm);
      const float scl = __expf(mrow[r] - mn);
      const float e0 = __expf(p[0][r] - mn);
      const float e1 = __expf(p[1][r] - mn);
      p[0][r] = e0;
      p[1][r] = e1;
      float rs = e0 + e1;
#pragma unroll
      for (int off = 1; off < 16; off <<= 1) rs += __shfl_xor(rs, off, 16);
      lrow[r] = lrow[r] * scl + rs;
      mrow[r] = mn;
      sf[r] = scl;
    }
#pragma unroll
    for (int dt = 0; dt < 4; ++dt)
#pragma unroll
      for (int r = 0; r < 8; ++r) o[dt][r] *= sf[r];

    // P: C-layout -> LDS -> A-layout (same-wave LDS ops are in-order)
#pragma unroll
    for (int r = 0; r < 8; ++r) {
      Ps[wave][r + 8 * half][laneM] = (__bf16)p[0][r];
      Ps[wave][r + 8 * half][16 + laneM] = (__bf16)p[1][r];
    }
    bf16x16 ap = cat8(lds_ld8(&Ps[wave][laneM][16 * half]),
                      lds_ld8(&Ps[wave][laneM][16 * half + 8]));
#pragma unroll
    for (int dt = 0; dt < 4; ++dt) {
      bf16x16 bv = cat8(lds_ld8(&Vt[dt * 16 + laneM][16 * half]),
                        lds_ld8(&Vt[dt * 16 + laneM][16 * half + 8]));
      o[dt] = wmma_bf16(ap, bv, o[dt]);
    }
  }

  // Normalize and write back to interleaved [B,T,C] bf16
  const int Cdim = H * 64;
#pragma unroll
  for (int dt = 0; dt < 4; ++dt) {
#pragma unroll
    for (int r = 0; r < 8; ++r) {
      const int t = qt + r + 8 * half;
      const float v = o[dt][r] / lrow[r];
      Y[(size_t)(b * T + t) * Cdim + h * 64 + dt * 16 + laneM] = (__bf16)v;
    }
  }
}

// ---------------------------------------------------------------------------
// Host-side orchestration
// ---------------------------------------------------------------------------
extern "C" void kernel_launch(void* const* d_in, const int* in_sizes, int n_in,
                              void* d_out, int out_size, void* d_ws,
                              size_t ws_size, hipStream_t stream) {
  (void)in_sizes; (void)n_in; (void)out_size; (void)ws_size;
  const int B = 2, T = 2048, C = 1024, H = 16;
  const int M = B * T;  // 4096 rows

  const float* x      = (const float*)d_in[0];
  const float* ln1w   = (const float*)d_in[1];
  const float* wattn  = (const float*)d_in[2];
  const float* wproj  = (const float*)d_in[3];
  const float* ln2w   = (const float*)d_in[4];
  const float* wfc    = (const float*)d_in[5];
  const float* wcproj = (const float*)d_in[6];

  char* ws = (char*)d_ws;
  size_t off = 0;
  auto alloc = [&](size_t bytes) -> void* {
    void* p = ws + off;
    off = (off + bytes + 255) & ~(size_t)255;
    return p;
  };
  __bf16* wa_bf  = (__bf16*)alloc((size_t)C * 3 * C * 2);
  __bf16* wp_bf  = (__bf16*)alloc((size_t)C * C * 2);
  __bf16* wf_bf  = (__bf16*)alloc((size_t)C * 4 * C * 2);
  __bf16* wc_bf  = (__bf16*)alloc((size_t)4 * C * C * 2);
  __bf16* h_bf   = (__bf16*)alloc((size_t)M * C * 2);
  __bf16* q_bf   = (__bf16*)alloc((size_t)M * C * 2);
  __bf16* k_bf   = (__bf16*)alloc((size_t)M * C * 2);
  __bf16* v_bf   = (__bf16*)alloc((size_t)M * C * 2);
  __bf16* y_bf   = (__bf16*)alloc((size_t)M * C * 2);
  float*  x1     = (float*)alloc((size_t)M * C * 4);
  __bf16* h2_bf  = (__bf16*)alloc((size_t)M * C * 2);
  __bf16* act_bf = (__bf16*)alloc((size_t)M * 4 * C * 2);

  auto cvt = [&](const float* src, __bf16* dst, int n) {
    f32_to_bf16_kernel<<<(n + 255) / 256, 256, 0, stream>>>(src, dst, n);
  };
  cvt(wattn, wa_bf, C * 3 * C);
  cvt(wproj, wp_bf, C * C);
  cvt(wfc, wf_bf, C * 4 * C);
  cvt(wcproj, wc_bf, 4 * C * C);

  // LN1 -> bf16
  layernorm_to_bf16<<<M, 256, 0, stream>>>(x, ln1w, h_bf, C);

  // QKV GEMM with per-head split epilogue
  gemm_bf16_wmma<3><<<dim3(3 * C / 128, M / 128), 256, 0, stream>>>(
      h_bf, wa_bf, M, 3 * C, C, nullptr, nullptr, nullptr, q_bf, k_bf, v_bf, T,
      C);

  // Causal attention
  attn_wmma<<<dim3(B * H, T / 64), 128, 0, stream>>>(q_bf, k_bf, v_bf, y_bf, T,
                                                     H);

  // x1 = x + y @ w_proj
  gemm_bf16_wmma<1><<<dim3(C / 128, M / 128), 256, 0, stream>>>(
      y_bf, wp_bf, M, C, C, x1, x, nullptr, nullptr, nullptr, nullptr, T, C);

  // LN2 -> bf16
  layernorm_to_bf16<<<M, 256, 0, stream>>>(x1, ln2w, h2_bf, C);

  // act = relu(h2 @ w_fc) -> bf16
  gemm_bf16_wmma<2><<<dim3(4 * C / 128, M / 128), 256, 0, stream>>>(
      h2_bf, wf_bf, M, 4 * C, C, nullptr, nullptr, act_bf, nullptr, nullptr,
      nullptr, T, C);

  // out = x1 + act @ w_cproj
  gemm_bf16_wmma<1><<<dim3(C / 128, M / 128), 256, 0, stream>>>(
      act_bf, wc_bf, M, C, 4 * C, (float*)d_out, x1, nullptr, nullptr, nullptr,
      nullptr, T, C);
}
